// AssociativeLIF_13211319403046
// MI455X (gfx1250) — compile-verified, matrix operand order checked by reading, not run on hardware
//
#include <hip/hip_runtime.h>
#include <hip/hip_bf16.h>

// Associative LIF forward scan, persistent-state kernel for gfx1250 (MI455X).
// State lives in VGPRs across all T steps; cluster mixing uses
// v_wmma_f32_16x16x32_f16 executed by wave 0 of each block.
// x is software-pipelined (register double buffer) so HBM latency hides
// behind the barrier/reduction/WMMA phase; outputs use non-temporal stores.

#define T_STEPS 32
#define BATCH   64
#define DDIM    4096
#define NCL     64
#define KPC     64                 // neurons per cluster = DDIM / NCL
#define BLOCK   512
#define NPT     (DDIM / BLOCK)     // 8 neurons per thread
#define V_RESET_F (-0.1f)
#define REFRAC_LEN 2

typedef __attribute__((ext_vector_type(16))) _Float16 v16h;
typedef __attribute__((ext_vector_type(8)))  float    v8f;

__global__ __launch_bounds__(BLOCK)
void assoc_lif_kernel(const float* __restrict__ x,          // (T,B,D)
                      const float* __restrict__ threshold,  // (D)
                      const float* __restrict__ beta_mem_raw,
                      const float* __restrict__ beta_syn_raw,
                      const float* __restrict__ nw,         // (NC,NC) raw
                      const float* __restrict__ gain,       // (NC)
                      float* __restrict__ spikes_out,       // (T,B,D)
                      float* __restrict__ v_out)            // (T,B,D)
{
    __shared__ _Float16 Wlds[NCL * NCL];  // sigmoid(nw), row-major [c][c']
    __shared__ float    part[BLOCK];      // per-thread spike partial sums
    __shared__ _Float16 cf16[NCL];        // cluster firing fraction (WMMA B operand)
    __shared__ float    ylds[NCL];        // mixed cascade per cluster

    const int tid = threadIdx.x;
    const int b   = blockIdx.x;
    const int c   = tid & (NCL - 1);      // cluster of every neuron this thread owns

    // ---- one-time setup ----
    const float bm = 1.0f / (1.0f + __expf(-beta_mem_raw[0]));
    const float bs = 1.0f / (1.0f + __expf(-beta_syn_raw[0]));
    const float g  = gain[c];

    for (int i = tid; i < NCL * NCL; i += BLOCK)
        Wlds[i] = (_Float16)(1.0f / (1.0f + __expf(-nw[i])));

    float v[NPT], isyn[NPT], th[NPT];
    int   refrac[NPT];
#pragma unroll
    for (int j = 0; j < NPT; ++j) {
        v[j] = 0.0f; isyn[j] = 0.0f; refrac[j] = 0;
        th[j] = threshold[tid + j * BLOCK];
    }

    // prime the x pipeline: step 0 values
    float xn[NPT];
    {
        const float* xrow0 = x + (size_t)b * DDIM;
#pragma unroll
        for (int j = 0; j < NPT; ++j) xn[j] = xrow0[tid + j * BLOCK];
    }
    __syncthreads();

    for (int t = 0; t < T_STEPS; ++t) {
        float*       srow = spikes_out + ((size_t)t * BATCH + b) * DDIM;
        float*       vrow = v_out      + ((size_t)t * BATCH + b) * DDIM;

        // consume current x, immediately issue next step's loads so HBM
        // latency overlaps the reduction/WMMA/barrier phase below
        float xc[NPT];
#pragma unroll
        for (int j = 0; j < NPT; ++j) xc[j] = xn[j];
        if (t + 1 < T_STEPS) {
            const float* xnext = x + ((size_t)(t + 1) * BATCH + b) * DDIM;
#pragma unroll
            for (int j = 0; j < NPT; ++j) xn[j] = xnext[tid + j * BLOCK];
        }

        float s[NPT];
        float psum = 0.0f;
#pragma unroll
        for (int j = 0; j < NPT; ++j) {
            isyn[j] = bs * isyn[j] + xc[j];
            v[j]    = (refrac[j] > 0) ? V_RESET_F
                                      : (bm * v[j] + (1.0f - bm) * isyn[j]);
            s[j]    = (v[j] >= th[j]) ? 1.0f : 0.0f;
            psum   += s[j];
        }
        part[tid] = psum;
        __syncthreads();

        // finish per-cluster reduction: 64 threads sum 8 partials each
        if (tid < NCL) {
            float acc = 0.0f;
#pragma unroll
            for (int w = 0; w < BLOCK / NCL; ++w) acc += part[tid + w * NCL];
            cf16[tid] = (_Float16)(acc * (1.0f / (float)KPC));
        }
        __syncthreads();

        // wave 0: y = Wsig * cf via V_WMMA_F32_16X16X32_F16 (EXEC all-ones in wave)
        if (tid < 32) {
            const int lane = tid;
            const int m = lane & 15;          // A-matrix row within 16-tile
            const int h = lane >> 4;          // lane half selects K sub-block

            // B operand: cf broadcast across N; two K-chunks of 32
            v16h bfr[2];
#pragma unroll
            for (int ch = 0; ch < 2; ++ch) {
#pragma unroll
                for (int i = 0; i < 16; ++i) {
                    const int K = i + 8 * h + ((i >> 3) << 3);   // 16-bit 16x32 layout
                    bfr[ch][i] = cf16[32 * ch + K];
                }
            }
#pragma unroll
            for (int ti = 0; ti < 4; ++ti) {  // 4 output row tiles of 16 clusters
                v8f acc = {};
#pragma unroll
                for (int ch = 0; ch < 2; ++ch) {
                    v16h afr;
#pragma unroll
                    for (int i = 0; i < 16; ++i) {
                        const int K = i + 8 * h + ((i >> 3) << 3);
                        afr[i] = Wlds[(ti * 16 + m) * NCL + 32 * ch + K];
                    }
                    acc = __builtin_amdgcn_wmma_f32_16x16x32_f16(
                        false, afr, false, bfr[ch], (short)0, acc, false, false);
                }
                if (m == 0) {                 // column N=0 of D holds the matvec result
#pragma unroll
                    for (int r = 0; r < 8; ++r)
                        ylds[ti * 16 + 8 * h + r] = acc[r];
                }
            }
        }
        __syncthreads();

        const float casc = ylds[c] * g;
#pragma unroll
        for (int j = 0; j < NPT; ++j) {
            const int d = tid + j * BLOCK;
            isyn[j] += casc;
            v[j]    -= s[j] * th[j];
            refrac[j] = (s[j] > 0.0f) ? REFRAC_LEN
                                      : ((refrac[j] > 0) ? refrac[j] - 1 : 0);
            __builtin_nontemporal_store(s[j], srow + d);   // write-only streams
            __builtin_nontemporal_store(v[j], vrow + d);
        }
        __syncthreads();   // protect part[]/cf16/ylds reuse next iteration
    }
}

extern "C" void kernel_launch(void* const* d_in, const int* in_sizes, int n_in,
                              void* d_out, int out_size, void* d_ws, size_t ws_size,
                              hipStream_t stream) {
    const float* x   = (const float*)d_in[0];   // current_in (T,B,D)
    const float* th  = (const float*)d_in[1];   // threshold (D)
    const float* bmr = (const float*)d_in[2];   // beta_mem_raw scalar
    const float* bsr = (const float*)d_in[3];   // beta_syn_raw scalar
    const float* nw  = (const float*)d_in[4];   // neighbor_weights (NC,NC)
    const float* gn  = (const float*)d_in[5];   // cluster_gain (NC)

    float* spikes = (float*)d_out;                                   // (T,B,D)
    float* vtr    = spikes + (size_t)T_STEPS * BATCH * DDIM;         // (T,B,D)

    assoc_lif_kernel<<<BATCH, BLOCK, 0, stream>>>(x, th, bmr, bsr, nw, gn,
                                                  spikes, vtr);
}
